// ResNet_6914897346971
// MI455X (gfx1250) — matrix operand order repaired and checked by reading, not compiled, not run on hardware
//
#include <hip/hip_runtime.h>

// ---------------------------------------------------------------------------
// MI455X (gfx1250) implementation.
//
// Roofline: ~70 GFLOP of GEMM vs ~2 GB of HBM traffic (output alone is 419MB).
// At 23.3 TB/s the memory floor is ~90us; fp32 math would be compute-bound, so
// all GEMMs run through V_WMMA_F32_16X16X32_F16 (f16 A/B, f32 accumulate),
// which pushes compute far under the memory floor. Masked BatchNorm requires
// global column stats between every GEMM -> 6 stage kernels, stats via
// LDS-reduced global float atomics.
// ---------------------------------------------------------------------------

typedef __attribute__((ext_vector_type(16))) _Float16 v16h;
typedef __attribute__((ext_vector_type(8)))  _Float16 v8h;
typedef __attribute__((ext_vector_type(4)))  _Float16 v4h;
typedef __attribute__((ext_vector_type(8)))  float    v8f;
typedef __attribute__((ext_vector_type(4)))  float    v4f;

#define NV     512
#define NROWS  (NV * NV)      // 262144
#define HDIM   128
#define NAA    20
#define NAA2   400
#define EPSBN  1e-5f

__device__ __forceinline__ float gelu_exact(float x) {
    return 0.5f * x * (1.0f + erff(x * 0.70710678118654752f));
}

// ---------------------------------------------------------------------------
// k_init: zero stats (5 sets of S[128],Q[128]) + mask count (1 float).
// ---------------------------------------------------------------------------
__global__ void k_init(float* statsBase, int n) {
    int t = blockIdx.x * blockDim.x + threadIdx.x;
    if (t < n) statsBase[t] = 0.0f;
}

// ---------------------------------------------------------------------------
// k_mask: mf[r] = triu(i<j) && 0<d<thresh ; atomic count into nbuf.
// ---------------------------------------------------------------------------
__global__ void k_mask(const float* __restrict__ d, const int* __restrict__ thresh,
                       float* __restrict__ mf, float* __restrict__ nbuf) {
    int r = blockIdx.x * blockDim.x + threadIdx.x;
    int i = r / NV, j = r % NV;
    float dv = d[r];
    float th = (float)(*thresh);
    float m = (i < j && dv > 0.0f && dv < th) ? 1.0f : 0.0f;
    mf[r] = m;
    __shared__ float red[256];
    red[threadIdx.x] = m;
    __syncthreads();
    for (int s = 128; s > 0; s >>= 1) {
        if (threadIdx.x < s) red[threadIdx.x] += red[threadIdx.x + s];
        __syncthreads();
    }
    if (threadIdx.x == 0) atomicAdd(nbuf, red[0]);
}

// ---------------------------------------------------------------------------
// k_prep: pack W[K=128][N=NT*16] (f32 row-major) into f16 WMMA B-fragment
// order: frag[((nt*4+kt)*32 + lane)*16 + e] = W[kt*32 + (lane/16)*8 + (e%8) + (e/8)*16][nt*16 + lane%16]
// (per CDNA5 16-bit B 32x16 lane layout).
// ---------------------------------------------------------------------------
__global__ void k_prep(const float* __restrict__ W, _Float16* __restrict__ dst, int NT) {
    int t = blockIdx.x * blockDim.x + threadIdx.x;
    int total = NT * 4 * 32 * 16;
    if (t >= total) return;
    int e  = t & 15;
    int ln = (t >> 4) & 31;
    int kt = (t >> 9) & 3;
    int nt = t >> 11;
    int K = kt * 32 + (ln >> 4) * 8 + (e & 7) + (e >> 3) * 16;
    int n = nt * 16 + (ln & 15);
    dst[t] = (_Float16)W[K * (NT * 16) + n];
}

// ---------------------------------------------------------------------------
// Stage kernel: per-wave 16-row tile.
//   in (f32 [nRows][128]) -> optional BN(stats Sin/Qin/n)+GELU -> f16 LDS tile
//   -> WMMA f32_16x16x32_f16 against packed weight frags, C init = bias
//   -> optional residual add, optional mask-zero, write out (f32 [nRows][OW])
//   -> optional masked column sum / sumsq accumulation into Sout/Qout.
// ---------------------------------------------------------------------------
template <int NTILES, bool DO_BN, bool RESID, bool STATS, bool MASKOUT>
__global__ __launch_bounds__(128)
void k_stage(const float* __restrict__ in, const float* __restrict__ resid,
             float* __restrict__ out, const _Float16* __restrict__ bfrag,
             const float* __restrict__ bias,
             const float* __restrict__ gamma, const float* __restrict__ beta,
             const float* __restrict__ Sin, const float* __restrict__ Qin,
             float* __restrict__ Sout, float* __restrict__ Qout,
             const float* __restrict__ nbuf, const float* __restrict__ mf,
             int nRows) {
    constexpr int OW = NTILES * 16;

    __shared__ _Float16 atile[4][16][HDIM];   // 16KB: per-wave f16 A tiles
    __shared__ float    mfT[4][16];
    __shared__ float    sred0[HDIM];
    __shared__ float    sred1[HDIM];

    const int wave = threadIdx.x >> 5;
    const int lane = threadIdx.x & 31;
    const int waveGlobal = blockIdx.x * 4 + wave;
    const int totalWaves = gridDim.x * 4;
    const int totalTiles = nRows / 16;

    if (STATS) {
        if (threadIdx.x < HDIM) { sred0[threadIdx.x] = 0.0f; sred1[threadIdx.x] = 0.0f; }
    }

    // Per-lane BN constants for columns lane*4 .. lane*4+3.
    v4f mean_v = {0,0,0,0}, rstd_v = {0,0,0,0}, g_v = {0,0,0,0}, b_v = {0,0,0,0};
    if constexpr (DO_BN) {
        const int c0 = lane * 4;
        const float nM = *nbuf;
        v4f S4 = *(const v4f*)(Sin + c0);
        v4f Q4 = *(const v4f*)(Qin + c0);
        g_v = *(const v4f*)(gamma + c0);
        b_v = *(const v4f*)(beta  + c0);
        for (int i = 0; i < 4; ++i) {
            float mu  = S4[i] / nM;
            float var = Q4[i] / nM - mu * mu;
            mean_v[i] = mu;
            rstd_v[i] = rsqrtf(var + EPSBN);
        }
    }

    float accS[STATS ? NTILES : 1] = {};
    float accQ[STATS ? NTILES : 1] = {};

    const int rA = lane & 15;        // A-matrix row for this lane
    const int hi = lane >> 4;        // lane half
    const _Float16* arow = &atile[wave][rA][hi * 8];

    for (int tile = waveGlobal; tile < totalTiles; tile += totalWaves) {
        __syncthreads();   // protect LDS reuse across iterations
        const int rowbase = tile * 16;

        if (lane < 16) mfT[wave][lane] = mf[rowbase + lane];

        // Stage 16 rows: load, transform, convert to f16, store to LDS.
        for (int r = 0; r < 16; ++r) {
            v4f xv = *(const v4f*)(in + (size_t)(rowbase + r) * HDIM + lane * 4);
            if constexpr (DO_BN) {
                for (int i = 0; i < 4; ++i) {
                    float y = (xv[i] - mean_v[i]) * rstd_v[i] * g_v[i] + b_v[i];
                    xv[i] = gelu_exact(y);
                }
            }
            v4h hv = { (_Float16)xv[0], (_Float16)xv[1], (_Float16)xv[2], (_Float16)xv[3] };
            *(v4h*)&atile[wave][r][lane * 4] = hv;
        }
        __syncthreads();

        for (int nt = 0; nt < NTILES; ++nt) {
            const int col = nt * 16 + (lane & 15);
            const float bia = bias[col];
            v8f c = { bia, bia, bia, bia, bia, bia, bia, bia };

            #pragma unroll
            for (int kt = 0; kt < 4; ++kt) {
                v8h lo = *(const v8h*)(arow + kt * 32);
                v8h hv = *(const v8h*)(arow + kt * 32 + 16);
                v16h a = __builtin_shufflevector(lo, hv,
                    0, 1, 2, 3, 4, 5, 6, 7, 8, 9, 10, 11, 12, 13, 14, 15);
                v16h bf = *(const v16h*)(bfrag + ((size_t)(nt * 4 + kt) * 32 + lane) * 16);
                c = __builtin_amdgcn_wmma_f32_16x16x32_f16(
                        false, a, false, bf, (short)0, c, false, false);
            }

            #pragma unroll
            for (int v = 0; v < 8; ++v) {
                const int lr  = v + 8 * hi;
                const int row = rowbase + lr;
                float val = c[v];
                if constexpr (RESID)   val += resid[(size_t)row * HDIM + col];
                if constexpr (MASKOUT) val *= mfT[wave][lr];
                out[(size_t)row * OW + col] = val;
                if constexpr (STATS) {
                    const float m = mfT[wave][lr];
                    accS[nt] += m * val;
                    accQ[nt] += m * val * val;
                }
            }
        }
    }

    if constexpr (STATS) {
        __syncthreads();
        for (int nt = 0; nt < NTILES; ++nt) {
            const int col = nt * 16 + (lane & 15);
            atomicAdd(&sred0[col], accS[nt]);
            atomicAdd(&sred1[col], accQ[nt]);
        }
        __syncthreads();
        if (threadIdx.x < HDIM) {
            atomicAdd(&Sout[threadIdx.x], sred0[threadIdx.x]);
            atomicAdd(&Qout[threadIdx.x], sred1[threadIdx.x]);
        }
    }
}

// ---------------------------------------------------------------------------
// k_sym: out[j,i,a,b] = out[i,j,b,a] for i<j (P[j,i]==0 since mask ⊂ triu;
// diagonal rows already zero from mask). In-place on d_out; each block owns
// one unordered pair so there are no read/write conflicts.
// ---------------------------------------------------------------------------
__global__ void k_sym(float* __restrict__ outp) {
    const int i = blockIdx.x, j = blockIdx.y;
    if (i >= j) return;
    __shared__ float rowBuf[NAA2];
    const float* src = outp + (size_t)(i * NV + j) * NAA2;
    for (int c = threadIdx.x; c < NAA2; c += blockDim.x) rowBuf[c] = src[c];
    __syncthreads();
    float* dst = outp + (size_t)(j * NV + i) * NAA2;
    for (int c = threadIdx.x; c < NAA2; c += blockDim.x) {
        int a = c / NAA, b = c % NAA;
        dst[c] = rowBuf[b * NAA + a];
    }
}

// ---------------------------------------------------------------------------
// Host-side orchestration.
// ---------------------------------------------------------------------------
extern "C" void kernel_launch(void* const* d_in, const int* in_sizes, int n_in,
                              void* d_out, int out_size, void* d_ws, size_t ws_size,
                              hipStream_t stream) {
    const float* x        = (const float*)d_in[0];
    const float* d        = (const float*)d_in[1];
    const float* W_in     = (const float*)d_in[2];
    const float* b_in     = (const float*)d_in[3];
    const float* blk_bn_g = (const float*)d_in[4];   // [2][2][128]
    const float* blk_bn_b = (const float*)d_in[5];
    const float* blk_W    = (const float*)d_in[6];   // [2][2][128][128]
    const float* blk_b    = (const float*)d_in[7];   // [2][2][128]
    const float* fin_g    = (const float*)d_in[8];
    const float* fin_b    = (const float*)d_in[9];
    const float* W_out    = (const float*)d_in[10];  // [128][400]
    const float* b_out    = (const float*)d_in[11];
    const int*   thresh   = (const int*)d_in[12];
    float* outp = (float*)d_out;

    // Workspace layout (~270MB needed):
    char* ws = (char*)d_ws;
    float* mf     = (float*)(ws);                              // 1MB
    float* statsF = (float*)(ws + (1u << 20));                 // 5*(128+128)+1 floats
    float* nbuf   = statsF + 5 * 256;
    _Float16* fWin  = (_Float16*)(ws + (1u << 20) + 8192);     // 32KB
    _Float16* fW00  = fWin  + 8 * 4 * 32 * 16;
    _Float16* fW01  = fW00  + 8 * 4 * 32 * 16;
    _Float16* fW10  = fW01  + 8 * 4 * 32 * 16;
    _Float16* fW11  = fW10  + 8 * 4 * 32 * 16;
    _Float16* fWout = fW11  + 8 * 4 * 32 * 16;                 // 100KB
    float* Hbuf = (float*)(ws + (size_t)(2u << 20));                       // 128MB
    float* Tbuf = (float*)(ws + (size_t)(2u << 20) + (size_t)NROWS * HDIM * 4);

    float* S0 = statsF + 0 * 256; float* Q0 = S0 + 128;
    float* S1 = statsF + 1 * 256; float* Q1 = S1 + 128;
    float* S2 = statsF + 2 * 256; float* Q2 = S2 + 128;
    float* S3 = statsF + 3 * 256; float* Q3 = S3 + 128;
    float* S4 = statsF + 4 * 256; float* Q4 = S4 + 128;

    // 1) zero stats + mask count
    k_init<<<6, 256, 0, stream>>>(statsF, 5 * 256 + 1);
    // 2) mask + n
    k_mask<<<NROWS / 256, 256, 0, stream>>>(d, thresh, mf, nbuf);
    // 3) pack weights into WMMA B-fragment order (f16)
    k_prep<<<64, 256, 0, stream>>>(W_in,                  fWin, 8);
    k_prep<<<64, 256, 0, stream>>>(blk_W + 0 * 128 * 128, fW00, 8);
    k_prep<<<64, 256, 0, stream>>>(blk_W + 1 * 128 * 128, fW01, 8);
    k_prep<<<64, 256, 0, stream>>>(blk_W + 2 * 128 * 128, fW10, 8);
    k_prep<<<64, 256, 0, stream>>>(blk_W + 3 * 128 * 128, fW11, 8);
    k_prep<<<200, 256, 0, stream>>>(W_out,                fWout, 25);

    const dim3 sg(1024), sb(128);
    // S0: h0 = x @ W_in + b_in ; stats(h0) -> set0
    k_stage<8, false, false, true, false><<<sg, sb, 0, stream>>>(
        x, nullptr, Hbuf, fWin, b_in, nullptr, nullptr,
        nullptr, nullptr, S0, Q0, nbuf, mf, NROWS);
    // S1: t0 = gelu(bn(h0; s0, g00)) @ W00 + b00 ; stats(t0) -> set1
    k_stage<8, true, false, true, false><<<sg, sb, 0, stream>>>(
        Hbuf, nullptr, Tbuf, fW00, blk_b + 0, blk_bn_g + 0, blk_bn_b + 0,
        S0, Q0, S1, Q1, nbuf, mf, NROWS);
    // S2: h1 = h0 + gelu(bn(t0; s1, g01)) @ W01 + b01 ; stats(h1) -> set2
    k_stage<8, true, true, true, false><<<sg, sb, 0, stream>>>(
        Tbuf, Hbuf, Hbuf, fW01, blk_b + 128, blk_bn_g + 128, blk_bn_b + 128,
        S1, Q1, S2, Q2, nbuf, mf, NROWS);
    // S3: t1 = gelu(bn(h1; s2, g10)) @ W10 + b10 ; stats(t1) -> set3
    k_stage<8, true, false, true, false><<<sg, sb, 0, stream>>>(
        Hbuf, nullptr, Tbuf, fW10, blk_b + 256, blk_bn_g + 256, blk_bn_b + 256,
        S2, Q2, S3, Q3, nbuf, mf, NROWS);
    // S4: h2 = h1 + gelu(bn(t1; s3, g11)) @ W11 + b11 ; stats(h2) -> set4
    k_stage<8, true, true, true, false><<<sg, sb, 0, stream>>>(
        Tbuf, Hbuf, Hbuf, fW11, blk_b + 384, blk_bn_g + 384, blk_bn_b + 384,
        S3, Q3, S4, Q4, nbuf, mf, NROWS);
    // S5: pred = gelu(bn(h2; s4, fin)) @ W_out + b_out, masked -> d_out
    k_stage<25, true, false, false, true><<<sg, sb, 0, stream>>>(
        Hbuf, nullptr, outp, fWout, b_out, fin_g, fin_b,
        S4, Q4, nullptr, nullptr, nbuf, mf, NROWS);
    // S6: symmetrize in place: out[j,i] = inner-transpose(out[i,j]) for i<j
    k_sym<<<dim3(NV, NV), 128, 0, stream>>>(outp);
}
